// NN_Model_51874615001736
// MI455X (gfx1250) — matrix operand order repaired and checked by reading, not compiled, not run on hardware
//
#include <hip/hip_runtime.h>

#define NG     16
#define NMOL   1024
#define NPRO   4096
#define NNODES 5120
#define HID    128
#define MAXE   250000
#define GNODES 320   // 64 mol + 256 pro per graph
#define EB     128   // edges per block in edge_mlp_wmma

typedef unsigned int uint;
typedef __attribute__((ext_vector_type(16))) __bf16 bf16x16;
typedef __attribute__((ext_vector_type(8)))  float  floatx8;

union V16 { bf16x16 v; uint u[8]; };

__device__ __forceinline__ float siluf(float x){ return x / (1.0f + __expf(-x)); }
__device__ __forceinline__ uint f2bf(float f){
  uint u = __float_as_uint(f);
  return ((u + 0x7FFFu + ((u >> 16) & 1u)) >> 16) & 0xFFFFu;
}
__device__ __forceinline__ uint pack2(float a, float b){ return f2bf(a) | (f2bf(b) << 16); }

// ---------------------------------------------------------------- utilities
__global__ void zero_f(float* p, int n){
  int i = blockIdx.x * blockDim.x + threadIdx.x;
  if (i < n) p[i] = 0.0f;
}
__global__ void zero_i(int* p, int n){
  int i = blockIdx.x * blockDim.x + threadIdx.x;
  if (i < n) p[i] = 0;
}

// ---------------------------------------------------------------- encoders + feature build
// feat[i][0:3]=x, [3:35]=encoder(32), [35:45]=PE(10), [45]=t
__global__ __launch_bounds__(256) void encode_kernel(
    const float* __restrict__ zm, const float* __restrict__ zp, const float* __restrict__ tv,
    const int* __restrict__ midx, const int* __restrict__ pidx, const int* __restrict__ mpos,
    const float* __restrict__ aw1, const float* __restrict__ ab1,
    const float* __restrict__ aw2, const float* __restrict__ ab2,
    const float* __restrict__ rw1, const float* __restrict__ rb1,
    const float* __restrict__ rw2, const float* __restrict__ rb2,
    float* __restrict__ feat)
{
  int i = blockIdx.x * blockDim.x + threadIdx.x;
  if (i >= NNODES) return;
  float* f = feat + (size_t)i * 46;
  if (i < NMOL) {
    const float* z = zm + (size_t)i * 19;
    f[0] = z[0]; f[1] = z[1]; f[2] = z[2];
    float hid[32];
    #pragma unroll
    for (int j = 0; j < 32; ++j){
      float s = ab1[j];
      #pragma unroll
      for (int k = 0; k < 16; ++k) s += z[3 + k] * aw1[k * 32 + j];
      hid[j] = siluf(s);
    }
    #pragma unroll
    for (int j = 0; j < 32; ++j){
      float s = ab2[j];
      #pragma unroll
      for (int k = 0; k < 32; ++k) s += hid[k] * aw2[k * 32 + j];
      f[3 + j] = s;
    }
    float pos = (float)mpos[i];
    #pragma unroll
    for (int j = 0; j < 5; ++j){
      float fr  = powf(10000.0f, -2.0f * (float)j / 10.0f);
      float ang = pos * fr;
      f[35 + j] = sinf(ang);
      f[40 + j] = cosf(ang);
    }
    f[45] = tv[midx[i]];
  } else {
    int ip = i - NMOL;
    const float* z = zp + (size_t)ip * 23;
    f[0] = z[0]; f[1] = z[1]; f[2] = z[2];
    float hid[40];
    #pragma unroll
    for (int j = 0; j < 40; ++j){
      float s = rb1[j];
      #pragma unroll
      for (int k = 0; k < 20; ++k) s += z[3 + k] * rw1[k * 40 + j];
      hid[j] = siluf(s);
    }
    #pragma unroll
    for (int j = 0; j < 32; ++j){
      float s = rb2[j];
      #pragma unroll
      for (int k = 0; k < 40; ++k) s += hid[k] * rw2[k * 32 + j];
      f[3 + j] = s;
    }
    #pragma unroll
    for (int j = 0; j < 10; ++j) f[35 + j] = 0.0f;
    f[45] = tv[pidx[ip]];
  }
}

// h0 = feat(5120x46) @ gnn_in_w(46x128) + b  (tiny; scalar)
__global__ __launch_bounds__(256) void gemm_in_kernel(
    const float* __restrict__ feat, const float* __restrict__ W,
    const float* __restrict__ b, float* __restrict__ h)
{
  int idx = blockIdx.x * 256 + threadIdx.x;     // exactly 5120*128
  int i = idx >> 7, j = idx & 127;
  const float* f = feat + (size_t)i * 46;
  float s = b[j];
  #pragma unroll
  for (int k = 0; k < 46; ++k) s += f[k] * W[k * 128 + j];
  h[(size_t)i * 128 + j] = s;
}

// ---------------------------------------------------------------- edge list (same graph, d^2 <= 1, incl self-loops)
__global__ __launch_bounds__(256) void build_edges(
    const float* __restrict__ zm, const float* __restrict__ zp,
    int* __restrict__ row, int* __restrict__ col, int* __restrict__ ety,
    int* __restrict__ count)
{
  int idx = blockIdx.x * 256 + threadIdx.x;     // exactly 16*320*320
  if (idx >= NG * GNODES * GNODES) return;
  int g   = idx / (GNODES * GNODES);
  int rem = idx % (GNODES * GNODES);
  int li = rem / GNODES, lj = rem % GNODES;
  float xi[3], xj[3];
  int gi, gj;
  if (li < 64){ gi = g * 64 + li; const float* z = zm + (size_t)gi * 19; xi[0]=z[0]; xi[1]=z[1]; xi[2]=z[2]; }
  else { int pp = g * 256 + (li - 64); gi = NMOL + pp; const float* z = zp + (size_t)pp * 23; xi[0]=z[0]; xi[1]=z[1]; xi[2]=z[2]; }
  if (lj < 64){ gj = g * 64 + lj; const float* z = zm + (size_t)gj * 19; xj[0]=z[0]; xj[1]=z[1]; xj[2]=z[2]; }
  else { int pp = g * 256 + (lj - 64); gj = NMOL + pp; const float* z = zp + (size_t)pp * 23; xj[0]=z[0]; xj[1]=z[1]; xj[2]=z[2]; }
  float dx = xi[0]-xj[0], dy = xi[1]-xj[1], dz = xi[2]-xj[2];
  float d2 = dx*dx + dy*dy + dz*dz;
  if (d2 <= 1.0f){
    int slot = atomicAdd(count, 1);
    if (slot < MAXE){
      row[slot] = gi; col[slot] = gj;
      ety[slot] = (li < 64 && lj < 64) ? 1 : ((li >= 64 && lj >= 64) ? 2 : 0);
    }
  }
}

// etb[l][t][j] = edge_b1[l][j] + sum_k edge_emb[t][k]*edge_w1[l][256+k][j]
__global__ void etb_kernel(const float* __restrict__ eemb, const float* __restrict__ ew1,
                           const float* __restrict__ eb1, float* __restrict__ etbb)
{
  int idx = blockIdx.x * 256 + threadIdx.x;
  if (idx >= 4 * 3 * 128) return;
  int l = idx / 384, t = (idx / 128) % 3, j = idx & 127;
  float s = eb1[l * 128 + j];
  const float* W = ew1 + (size_t)l * 264 * 128 + 256 * 128;
  #pragma unroll
  for (int k = 0; k < 8; ++k) s += eemb[t * 8 + k] * W[k * 128 + j];
  etbb[(l * 3 + t) * 128 + j] = s;
}

// ---------------------------------------------------------------- generic WMMA GEMM (bf16 in, f32 acc)
// OUT(nx128) = act( [A0|A1](nxK) @ W(Kx128) + bias ), optional residual add into OUT.
// block = 256 threads (8 waves), 32 rows/block, grid = 5120/32 = 160.
template<int K, bool DO_SILU, bool RESID>
__global__ __launch_bounds__(256) void gemm_wmma(
    const float* __restrict__ A0, const float* __restrict__ A1,
    const float* __restrict__ W,  const float* __restrict__ bias,
    float* __restrict__ out)
{
  constexpr int KP = K / 2;       // bf16 pairs per row
  constexpr int LP = KP + 1;      // padded LDS stride (bank-conflict avoidance)
  __shared__ uint Wt[128 * LP];   // Wt[col][kpair]  (transposed weights)
  __shared__ uint Ab[32 * LP];    // Ab[rowLocal][kpair]
  int tid = threadIdx.x;
  int rowBase = blockIdx.x * 32;

  for (int idx = tid; idx < 128 * KP; idx += 256){
    int j = idx / KP, kk = idx % KP;
    Wt[j * LP + kk] = pack2(W[(2 * kk) * 128 + j], W[(2 * kk + 1) * 128 + j]);
  }
  for (int idx = tid; idx < 32 * KP; idx += 256){
    int e = idx / KP, kk = idx % KP;
    size_t r = (size_t)(rowBase + e) * 128;
    float f0, f1;
    if (K == 256 && kk >= 64){ f0 = A1[r + 2 * kk - 128]; f1 = A1[r + 2 * kk - 127]; }
    else                      { f0 = A0[r + 2 * kk];       f1 = A0[r + 2 * kk + 1];   }
    Ab[e * LP + kk] = pack2(f0, f1);
  }
  __syncthreads();

  int wave = tid >> 5, lane = tid & 31, half = lane >> 4, ln = lane & 15;
  int mt = wave >> 2, np = wave & 3;            // 2 row-tiles x 4 col-pair groups
  int c0 = (np * 2 + 0) * 16 + ln;
  int c1 = (np * 2 + 1) * 16 + ln;
  floatx8 acc0, acc1;
  {
    float b0v = bias ? bias[c0] : 0.0f;
    float b1v = bias ? bias[c1] : 0.0f;
    #pragma unroll
    for (int v = 0; v < 8; ++v){ acc0[v] = b0v; acc1[v] = b1v; }
  }
  int m = mt * 16 + ln;
  #pragma unroll
  for (int ks = 0; ks < K / 32; ++ks){
    V16 a;
    int a0 = 16 * ks + 8 * half, a1 = 16 * ks + 8 * half + 4;
    #pragma unroll
    for (int i2 = 0; i2 < 4; ++i2){ a.u[i2] = Ab[m * LP + a0 + i2]; a.u[4 + i2] = Ab[m * LP + a1 + i2]; }
    int bb = 16 * ks + 8 * half;
    V16 b;
    #pragma unroll
    for (int i2 = 0; i2 < 8; ++i2) b.u[i2] = Wt[c0 * LP + bb + i2];
    acc0 = __builtin_amdgcn_wmma_f32_16x16x32_bf16(false, a.v, false, b.v, (short)0, acc0, false, false);
    #pragma unroll
    for (int i2 = 0; i2 < 8; ++i2) b.u[i2] = Wt[c1 * LP + bb + i2];
    acc1 = __builtin_amdgcn_wmma_f32_16x16x32_bf16(false, a.v, false, b.v, (short)0, acc1, false, false);
  }
  #pragma unroll
  for (int v = 0; v < 8; ++v){
    size_t r = (size_t)(rowBase + mt * 16 + v + 8 * half) * 128;
    float v0 = acc0[v], v1 = acc1[v];
    if (DO_SILU){ v0 = siluf(v0); v1 = siluf(v1); }
    if (RESID){ out[r + c0] += v0; out[r + c1] += v1; }
    else      { out[r + c0]  = v0; out[r + c1]  = v1; }
  }
}

// ---------------------------------------------------------------- edge second layer (WMMA) + segment-sum
// A row = silu(p[row]+q[col]+etb[et]); out = silu(A@W2+b2)/NORM atomically into agg[row].
// 128 edges/block, 8 waves; wave w owns edge tile w (16 edges), caches its 4 A
// K-fragments in registers, loops over all 8 N-tiles (2 interleaved acc chains).
__global__ __launch_bounds__(256) void edge_mlp_wmma(
    const float* __restrict__ p, const float* __restrict__ q, const float* __restrict__ etb,
    const int* __restrict__ row, const int* __restrict__ col, const int* __restrict__ ety,
    const int* __restrict__ countPtr,
    const float* __restrict__ W2, const float* __restrict__ b2,
    float* __restrict__ agg)
{
  __shared__ uint Wt[128 * 65];   // W2 transposed, bf16 pairs: Wt[col][kpair]
  __shared__ uint Ab[EB * 65];    // edge activations, bf16 pairs
  __shared__ int  sr[EB], sc[EB], st[EB];
  int E = *countPtr; if (E > MAXE) E = MAXE;
  int base = blockIdx.x * EB;
  if (base >= E) return;                         // uniform early exit
  int tid = threadIdx.x;
  for (int idx = tid; idx < EB; idx += 256){
    int e = base + idx; if (e >= E) e = base;    // clamp to a valid edge for safe gathers
    sr[idx] = row[e]; sc[idx] = col[e]; st[idx] = ety[e];
  }
  for (int idx = tid; idx < 128 * 64; idx += 256){
    int j = idx >> 6, kk = idx & 63;
    Wt[j * 65 + kk] = pack2(W2[(2 * kk) * 128 + j], W2[(2 * kk + 1) * 128 + j]);
  }
  __syncthreads();
  for (int idx = tid; idx < EB * 64; idx += 256){
    int e = idx >> 6, kk = idx & 63;
    size_t r = (size_t)sr[e] * 128, c = (size_t)sc[e] * 128;
    int tb = st[e] * 128, k0 = 2 * kk;
    float f0 = siluf(p[r + k0]     + q[c + k0]     + etb[tb + k0]);
    float f1 = siluf(p[r + k0 + 1] + q[c + k0 + 1] + etb[tb + k0 + 1]);
    Ab[e * 65 + kk] = pack2(f0, f1);
  }
  __syncthreads();

  int wave = tid >> 5, lane = tid & 31, half = lane >> 4, ln = lane & 15;
  int m = wave * 16 + ln;                        // wave owns 16-edge tile 'wave'

  // cache all 4 K-step A fragments in registers (reused by all 8 N-tiles)
  V16 afrag[4];
  #pragma unroll
  for (int ks = 0; ks < 4; ++ks){
    int a0 = 16 * ks + 8 * half, a1 = 16 * ks + 8 * half + 4;
    #pragma unroll
    for (int i2 = 0; i2 < 4; ++i2){
      afrag[ks].u[i2]     = Ab[m * 65 + a0 + i2];
      afrag[ks].u[4 + i2] = Ab[m * 65 + a1 + i2];
    }
  }

  #pragma unroll
  for (int np = 0; np < 4; ++np){
    int c0 = (np * 2 + 0) * 16 + ln;
    int c1 = (np * 2 + 1) * 16 + ln;
    floatx8 acc0, acc1;
    {
      float b0v = b2[c0], b1v = b2[c1];
      #pragma unroll
      for (int v = 0; v < 8; ++v){ acc0[v] = b0v; acc1[v] = b1v; }
    }
    #pragma unroll
    for (int ks = 0; ks < 4; ++ks){
      int bb = 16 * ks + 8 * half;
      V16 b;
      #pragma unroll
      for (int i2 = 0; i2 < 8; ++i2) b.u[i2] = Wt[c0 * 65 + bb + i2];
      acc0 = __builtin_amdgcn_wmma_f32_16x16x32_bf16(false, afrag[ks].v, false, b.v, (short)0, acc0, false, false);
      #pragma unroll
      for (int i2 = 0; i2 < 8; ++i2) b.u[i2] = Wt[c1 * 65 + bb + i2];
      acc1 = __builtin_amdgcn_wmma_f32_16x16x32_bf16(false, afrag[ks].v, false, b.v, (short)0, acc1, false, false);
    }
    #pragma unroll
    for (int v = 0; v < 8; ++v){
      int eLoc = wave * 16 + v + 8 * half;
      if (base + eLoc < E){
        float* dst = agg + (size_t)sr[eLoc] * 128;
        float v0 = siluf(acc0[v]) * 0.01f;      // /NORM
        float v1 = siluf(acc1[v]) * 0.01f;
        __hip_atomic_fetch_add(dst + c0, v0, __ATOMIC_RELAXED, __HIP_MEMORY_SCOPE_AGENT);
        __hip_atomic_fetch_add(dst + c1, v1, __ATOMIC_RELAXED, __HIP_MEMORY_SCOPE_AGENT);
      }
    }
  }
}

// ---------------------------------------------------------------- output head + decoders
__global__ __launch_bounds__(256) void decode_kernel(
    const float* __restrict__ h, const float* __restrict__ ow, const float* __restrict__ ob,
    const float* __restrict__ adw1, const float* __restrict__ adb1,
    const float* __restrict__ adw2, const float* __restrict__ adb2,
    const float* __restrict__ rdw1, const float* __restrict__ rdb1,
    const float* __restrict__ rdw2, const float* __restrict__ rdb2,
    float* __restrict__ out)
{
  int i = blockIdx.x * blockDim.x + threadIdx.x;
  if (i >= NNODES) return;
  float o[46];
  #pragma unroll
  for (int j = 0; j < 46; ++j) o[j] = ob[j];
  const float* hr = h + (size_t)i * 128;
  for (int k = 0; k < 128; ++k){
    float hv = hr[k];
    #pragma unroll
    for (int j = 0; j < 46; ++j) o[j] += hv * ow[k * 46 + j];
  }
  if (i < NMOL){
    float hid[32];
    #pragma unroll
    for (int j = 0; j < 32; ++j){
      float s = adb1[j];
      #pragma unroll
      for (int k = 0; k < 32; ++k) s += o[3 + k] * adw1[k * 32 + j];
      hid[j] = siluf(s);
    }
    float* po = out + (size_t)i * 19;
    po[0] = o[0]; po[1] = o[1]; po[2] = o[2];
    #pragma unroll
    for (int j = 0; j < 16; ++j){
      float s = adb2[j];
      #pragma unroll
      for (int k = 0; k < 32; ++k) s += hid[k] * adw2[k * 16 + j];
      po[3 + j] = s;
    }
  } else {
    int ip = i - NMOL;
    float hid[40];
    #pragma unroll
    for (int j = 0; j < 40; ++j){
      float s = rdb1[j];
      #pragma unroll
      for (int k = 0; k < 32; ++k) s += o[3 + k] * rdw1[k * 40 + j];
      hid[j] = siluf(s);
    }
    float* po = out + (size_t)NMOL * 19 + (size_t)ip * 23;
    po[0] = o[0]; po[1] = o[1]; po[2] = o[2];
    #pragma unroll
    for (int j = 0; j < 20; ++j){
      float s = rdb2[j];
      #pragma unroll
      for (int k = 0; k < 40; ++k) s += hid[k] * rdw2[k * 20 + j];
      po[3 + j] = s;
    }
  }
}

// ---------------------------------------------------------------- host
extern "C" void kernel_launch(void* const* d_in, const int* in_sizes, int n_in,
                              void* d_out, int out_size, void* d_ws, size_t ws_size,
                              hipStream_t stream)
{
  const float* zm   = (const float*)d_in[0];
  const float* zp   = (const float*)d_in[1];
  const float* tv   = (const float*)d_in[2];
  const int*   midx = (const int*)d_in[3];
  const int*   pidx = (const int*)d_in[4];
  const int*   mpos = (const int*)d_in[5];
  const float* aw1 = (const float*)d_in[6],  *ab1 = (const float*)d_in[7];
  const float* aw2 = (const float*)d_in[8],  *ab2 = (const float*)d_in[9];
  const float* rw1 = (const float*)d_in[10], *rb1 = (const float*)d_in[11];
  const float* rw2 = (const float*)d_in[12], *rb2 = (const float*)d_in[13];
  const float* adw1 = (const float*)d_in[14], *adb1 = (const float*)d_in[15];
  const float* adw2 = (const float*)d_in[16], *adb2 = (const float*)d_in[17];
  const float* rdw1 = (const float*)d_in[18], *rdb1 = (const float*)d_in[19];
  const float* rdw2 = (const float*)d_in[20], *rdb2 = (const float*)d_in[21];
  const float* eemb = (const float*)d_in[22];
  const float* giw = (const float*)d_in[23], *gib = (const float*)d_in[24];
  const float* gow = (const float*)d_in[25], *gob = (const float*)d_in[26];
  const float* ew1 = (const float*)d_in[27], *eb1 = (const float*)d_in[28];
  const float* ew2 = (const float*)d_in[29], *eb2 = (const float*)d_in[30];
  const float* nw1 = (const float*)d_in[31], *nb1 = (const float*)d_in[32];
  const float* nw2 = (const float*)d_in[33], *nb2 = (const float*)d_in[34];

  float* h    = (float*)d_ws;
  float* pbuf = h    + (size_t)NNODES * HID;
  float* qbuf = pbuf + (size_t)NNODES * HID;
  float* agg  = qbuf + (size_t)NNODES * HID;
  float* tbuf = agg  + (size_t)NNODES * HID;
  float* feat = tbuf + (size_t)NNODES * HID;
  float* etbb = feat + (size_t)NNODES * 46;
  float* fend = etbb + 4 * 3 * 128;
  int* rowA  = (int*)fend;
  int* colA  = rowA + MAXE;
  int* etyA  = colA + MAXE;
  int* count = etyA + MAXE;

  zero_i<<<1, 32, 0, stream>>>(count, 1);
  encode_kernel<<<(NNODES + 255) / 256, 256, 0, stream>>>(
      zm, zp, tv, midx, pidx, mpos, aw1, ab1, aw2, ab2, rw1, rb1, rw2, rb2, feat);
  gemm_in_kernel<<<(NNODES * HID) / 256, 256, 0, stream>>>(feat, giw, gib, h);
  build_edges<<<(NG * GNODES * GNODES) / 256, 256, 0, stream>>>(zm, zp, rowA, colA, etyA, count);
  etb_kernel<<<6, 256, 0, stream>>>(eemb, ew1, eb1, etbb);

  for (int l = 0; l < 4; ++l){
    const float* W1 = ew1 + (size_t)l * 264 * 128;
    gemm_wmma<128, false, false><<<NNODES / 32, 256, 0, stream>>>(h, nullptr, W1, nullptr, pbuf);
    gemm_wmma<128, false, false><<<NNODES / 32, 256, 0, stream>>>(h, nullptr, W1 + 128 * 128, nullptr, qbuf);
    zero_f<<<(NNODES * HID + 255) / 256, 256, 0, stream>>>(agg, NNODES * HID);
    edge_mlp_wmma<<<(MAXE + EB - 1) / EB, 256, 0, stream>>>(
        pbuf, qbuf, etbb + l * 3 * 128, rowA, colA, etyA, count,
        ew2 + (size_t)l * 128 * 128, eb2 + l * 128, agg);
    gemm_wmma<256, true, false><<<NNODES / 32, 256, 0, stream>>>(
        h, agg, nw1 + (size_t)l * 256 * 128, nb1 + l * 128, tbuf);
    gemm_wmma<128, false, true><<<NNODES / 32, 256, 0, stream>>>(
        tbuf, nullptr, nw2 + (size_t)l * 128 * 128, nb2 + l * 128, h);
  }

  decode_kernel<<<(NNODES + 255) / 256, 256, 0, stream>>>(
      h, gow, gob, adw1, adb1, adw2, adb2, rdw1, rdb1, rdw2, rdb2, (float*)d_out);
}
